// BatchTopKCrossCoder_66159676228078
// MI455X (gfx1250) — matrix operand order repaired and checked by reading, not compiled, not run on hardware
//
#include <hip/hip_runtime.h>
#include <hip/hip_bf16.h>

typedef __bf16 bf16_t;
typedef __attribute__((ext_vector_type(16))) __bf16 v16bf;
typedef __attribute__((ext_vector_type(8)))  float  v8f;
typedef __attribute__((ext_vector_type(4)))  int    v4i;

#define NBINS     8192
#define BIN_SHIFT 18
#define LMAX      1024

#if defined(__has_builtin)
#if __has_builtin(__builtin_amdgcn_global_load_async_to_lds_b128) && \
    __has_builtin(__builtin_amdgcn_s_wait_asynccnt)
#define HAVE_ASYNC_LDS 1
#endif
#endif

#if HAVE_ASYNC_LDS
typedef __attribute__((address_space(1))) v4i glb_v4i;
typedef __attribute__((address_space(3))) v4i lds_v4i;
#endif

union FragAB { v16bf v; uint4 q[2]; };

// ---------------------------------------------------------------------------
// Kernel 1: zero global histogram (ws is not re-poisoned between graph replays)
// ---------------------------------------------------------------------------
__global__ void cc_zero(unsigned* __restrict__ p, int n) {
  int i = blockIdx.x * blockDim.x + threadIdx.x;
  if (i < n) p[i] = 0u;
}

// ---------------------------------------------------------------------------
// Kernel 2: encode GEMM  acts[B,H] = relu(X[B,2D] * We[2D,H] + be)  (bf16 WMMA)
// Fused: per-block LDS histogram of positive activations -> global hist.
// Block = 128 threads (4 waves), tile 64(M) x 64(N), K-step 32.
// Epilogue bounces the C tile through LDS for coalesced b128 stores.
// ---------------------------------------------------------------------------
__global__ __launch_bounds__(128)
void cc_encode_wmma(const float* __restrict__ X,    // [B, 2D]
                    const float* __restrict__ We,   // [2D, H]
                    const float* __restrict__ be,   // [H]
                    float*       __restrict__ acts, // [B, H]
                    unsigned*    __restrict__ hist, // [NBINS]
                    int B, int H, int K2D)
{
  union ShEnc {
    struct { bf16_t a[64][32]; bf16_t b[64][32]; } s;  // staging (8KB)
    float cst[64][68];                                 // C bounce, 68 -> 16B rows
  };
  __shared__ __align__(16) ShEnc sh;
  __shared__ unsigned hlds[NBINS];

  const int tid  = threadIdx.x;
  const int wave = tid >> 5;
  const int lane = tid & 31;
  const int half = lane >> 4;      // 0: lanes 0-15, 1: lanes 16-31
  const int r    = lane & 15;
  const int m0   = wave * 16;      // wave's M base within the 64-row tile

  const int hx0 = blockIdx.x * 64; // N (feature) base
  const int bx0 = blockIdx.y * 64; // M (batch)   base

  // A staging: one 16-wide row segment per thread
  const int arow_s = tid >> 1;
  const int akc    = (tid & 1) * 16;
  // B staging: 4(K) x 4(N) sub-block per thread
  const int bkb = (tid >> 4) * 4;
  const int bnb = (tid & 15) * 4;

  for (int i = tid; i < NBINS; i += 128) hlds[i] = 0u;

  v8f acc[4] = {};

  for (int k0 = 0; k0 < K2D; k0 += 32) {
    // ---- stage A tile 64x32 f32 -> bf16, two b128 LDS stores per thread ----
    {
      const float* src = X + (size_t)(bx0 + arow_s) * K2D + k0 + akc;
      const float4 x0 = *(const float4*)(src + 0);
      const float4 x1 = *(const float4*)(src + 4);
      const float4 x2 = *(const float4*)(src + 8);
      const float4 x3 = *(const float4*)(src + 12);
      union { bf16_t h[8]; uint4 q; } pa, pb;
      pa.h[0] = (bf16_t)x0.x; pa.h[1] = (bf16_t)x0.y;
      pa.h[2] = (bf16_t)x0.z; pa.h[3] = (bf16_t)x0.w;
      pa.h[4] = (bf16_t)x1.x; pa.h[5] = (bf16_t)x1.y;
      pa.h[6] = (bf16_t)x1.z; pa.h[7] = (bf16_t)x1.w;
      pb.h[0] = (bf16_t)x2.x; pb.h[1] = (bf16_t)x2.y;
      pb.h[2] = (bf16_t)x2.z; pb.h[3] = (bf16_t)x2.w;
      pb.h[4] = (bf16_t)x3.x; pb.h[5] = (bf16_t)x3.y;
      pb.h[6] = (bf16_t)x3.z; pb.h[7] = (bf16_t)x3.w;
      *(uint4*)(&sh.s.a[arow_s][akc + 0]) = pa.q;
      *(uint4*)(&sh.s.a[arow_s][akc + 8]) = pb.q;
    }
    // ---- stage B tile 32(K)x64(N) f32 -> b[n][k], packed b64 stores ----
    {
      const float4 w0 = *(const float4*)(We + (size_t)(k0 + bkb + 0) * H + hx0 + bnb);
      const float4 w1 = *(const float4*)(We + (size_t)(k0 + bkb + 1) * H + hx0 + bnb);
      const float4 w2 = *(const float4*)(We + (size_t)(k0 + bkb + 2) * H + hx0 + bnb);
      const float4 w3 = *(const float4*)(We + (size_t)(k0 + bkb + 3) * H + hx0 + bnb);
      const float* f0 = &w0.x; const float* f1 = &w1.x;
      const float* f2 = &w2.x; const float* f3 = &w3.x;
      #pragma unroll
      for (int j = 0; j < 4; ++j) {
        union { bf16_t h[4]; uint2 u; } p;
        p.h[0] = (bf16_t)f0[j]; p.h[1] = (bf16_t)f1[j];
        p.h[2] = (bf16_t)f2[j]; p.h[3] = (bf16_t)f3[j];
        *(uint2*)(&sh.s.b[bnb + j][bkb]) = p.u;
      }
    }
    // prefetch next K-tile into the cache hierarchy (global_prefetch_b8)
    if (k0 + 32 < K2D) {
      __builtin_prefetch(X + (size_t)(bx0 + arow_s) * K2D + (k0 + 32) + akc);
      __builtin_prefetch(We + (size_t)(k0 + 32 + bkb) * H + hx0 + bnb);
    }
    __syncthreads();

    // ---- A fragment (ISA 16-bit A 16x32 layout):
    //   lanes 0-15  hold row M=r, K = 0..7  and 16..23
    //   lanes 16-31 hold row M=r, K = 8..15 and 24..31
    FragAB a;
    const uint4* arow = (const uint4*)(&sh.s.a[m0 + r][0]);
    a.q[0] = arow[half];       // K = half*8    .. +7
    a.q[1] = arow[2 + half];   // K = 16+half*8 .. +7

    // ---- B fragments (same K order as A: half 0 holds K 0..7,16..23; half 1 holds K 8..15,24..31)
    #pragma unroll
    for (int t = 0; t < 4; ++t) {
      FragAB bfr;
      const uint4* brow = (const uint4*)(&sh.s.b[t * 16 + r][0]);
      bfr.q[0] = brow[half];       // K = half*8    .. +7
      bfr.q[1] = brow[2 + half];   // K = 16+half*8 .. +7
      acc[t] = __builtin_amdgcn_wmma_f32_16x16x32_bf16(
          /*neg_a=*/false, a.v, /*neg_b=*/false, bfr.v,
          /*c_mod=*/(short)0, acc[t], /*reuse_a=*/false, /*reuse_b=*/false);
    }
    __syncthreads();
  }

  // ---- epilogue: bounce C tile through LDS, then coalesced bias+relu+store ----
  // C layout: VGPR i, lanes 0-15 -> (M=i, N=lane); lanes 16-31 -> (M=8+i, N=lane-16)
  #pragma unroll
  for (int t = 0; t < 4; ++t) {
    #pragma unroll
    for (int i = 0; i < 8; ++i) {
      sh.cst[m0 + half * 8 + i][t * 16 + r] = acc[t][i];
    }
  }
  __syncthreads();

  {
    const int rgrp = tid >> 4;          // 0..7  row within each group of 8
    const int col4 = (tid & 15) * 4;    // 16 float4 columns
    const float4 bb = *(const float4*)(be + hx0 + col4);
    #pragma unroll
    for (int rr = 0; rr < 8; ++rr) {
      const int row = rr * 8 + rgrp;    // 0..63
      float4 y = *(const float4*)(&sh.cst[row][col4]);
      y.x = fmaxf(y.x + bb.x, 0.0f);
      y.y = fmaxf(y.y + bb.y, 0.0f);
      y.z = fmaxf(y.z + bb.z, 0.0f);
      y.w = fmaxf(y.w + bb.w, 0.0f);
      if (y.x > 0.0f) atomicAdd(&hlds[__float_as_uint(y.x) >> BIN_SHIFT], 1u);
      if (y.y > 0.0f) atomicAdd(&hlds[__float_as_uint(y.y) >> BIN_SHIFT], 1u);
      if (y.z > 0.0f) atomicAdd(&hlds[__float_as_uint(y.z) >> BIN_SHIFT], 1u);
      if (y.w > 0.0f) atomicAdd(&hlds[__float_as_uint(y.w) >> BIN_SHIFT], 1u);
      *(float4*)(acts + (size_t)(bx0 + row) * H + hx0 + col4) = y;
    }
  }
  __syncthreads();
  for (int i = tid; i < NBINS; i += 128) {
    const unsigned c = hlds[i];
    if (c) atomicAdd(&hist[i], c);
  }
}

// ---------------------------------------------------------------------------
// Kernel 3: find threshold = lower edge of highest bin whose suffix-count >= k*B
// ---------------------------------------------------------------------------
__global__ __launch_bounds__(256)
void cc_threshold(const unsigned* __restrict__ hist,
                  const int*      __restrict__ kp,
                  float*          __restrict__ thr,
                  int B, int H)
{
  __shared__ unsigned part[256];
  __shared__ unsigned above[256];
  __shared__ unsigned totalAll;

  const int tid  = threadIdx.x;
  const int CH   = NBINS / 256;
  const int base = tid * CH;

  unsigned s = 0;
  for (int i = 0; i < CH; ++i) s += hist[base + i];
  part[tid] = s;
  __syncthreads();
  if (tid == 0) {
    unsigned run = 0;
    for (int i = 255; i >= 0; --i) { above[i] = run; run += part[i]; }
    totalAll = run;
  }
  __syncthreads();

  const unsigned long long kt  = (unsigned long long)(*kp) * (unsigned long long)B;
  const unsigned long long tot = (unsigned long long)B * (unsigned long long)H;
  const unsigned K_total = (unsigned)(kt < tot ? kt : tot);

  if (totalAll < K_total) {
    if (tid == 0) *thr = 0.0f;           // fewer positives than k*B: keep all
  } else if (above[tid] < K_total && above[tid] + part[tid] >= K_total) {
    unsigned run = above[tid];
    int bin = base;
    for (int i = CH - 1; i >= 0; --i) {
      run += hist[base + i];
      if (run >= K_total) { bin = base + i; break; }
    }
    *thr = __uint_as_float((unsigned)bin << BIN_SHIFT);
  }
}

// ---------------------------------------------------------------------------
// Kernel 4: sparse decode. One block per batch row: deterministic compaction,
// then out[b,:] = sum a_h * Wd[h,:] + bd with the active decoder rows pulled
// through LDS via GLOBAL_LOAD_ASYNC_TO_LDS_B128 double buffering (ASYNCcnt).
// ---------------------------------------------------------------------------
#define DEC_MAXRD 4096

__global__ __launch_bounds__(256)
void cc_decode(const float* __restrict__ acts, // [B, H]
               const float* __restrict__ Wd,   // [H, 2D]
               const float* __restrict__ bd,   // [2D]
               const float* __restrict__ thr,
               float*       __restrict__ out,  // [B, 2D]
               int H, int RD)
{
  __shared__ int   cnts[256];
  __shared__ int   sIdx[LMAX];
  __shared__ float sVal[LMAX];
  __shared__ int   nnz_s;
#if HAVE_ASYNC_LDS
  __shared__ __align__(16) float wbuf[2][DEC_MAXRD];   // 32KB double buffer
#endif

  const int b   = blockIdx.x;
  const int tid = threadIdx.x;
  const float th = *thr;
  const float* arow = acts + (size_t)b * H;

  // pass 1: count actives per thread (strided)
  int c = 0;
  for (int h = tid; h < H; h += 256) {
    const float v = arow[h];
    if (v >= th && v > 0.0f) ++c;
  }
  cnts[tid] = c;
  __syncthreads();
  if (tid == 0) {
    int run = 0;
    for (int i = 0; i < 256; ++i) { const int t = cnts[i]; cnts[i] = run; run += t; }
    nnz_s = run < LMAX ? run : LMAX;
  }
  __syncthreads();
  // pass 2: ordered compaction (deterministic)
  int o = cnts[tid];
  for (int h = tid; h < H; h += 256) {
    const float v = arow[h];
    if (v >= th && v > 0.0f) {
      if (o < LMAX) { sIdx[o] = h; sVal[o] = v; }
      ++o;
    }
  }
  __syncthreads();
  const int nnz = nnz_s;

#if HAVE_ASYNC_LDS
  if (RD <= DEC_MAXRD) {
    // async double-buffered gather of decoder rows through LDS
    float4 acc4[DEC_MAXRD / 1024];
    int nc = 0;
    for (int d = tid * 4; d < RD; d += 1024) acc4[nc++] = *(const float4*)(bd + d);

    if (nnz > 0) {
      // issue row 0
      {
        const float* src = Wd + (size_t)sIdx[0] * RD;
        for (int d = tid * 4; d < RD; d += 1024)
          __builtin_amdgcn_global_load_async_to_lds_b128(
              (glb_v4i*)(src + d), (lds_v4i*)(&wbuf[0][d]), 0, 0);
      }
      for (int i = 0; i < nnz; ++i) {
        const int buf = i & 1;
        if (i + 1 < nnz) {
          const float* src = Wd + (size_t)sIdx[i + 1] * RD;
          for (int d = tid * 4; d < RD; d += 1024)
            __builtin_amdgcn_global_load_async_to_lds_b128(
                (glb_v4i*)(src + d), (lds_v4i*)(&wbuf[buf ^ 1][d]), 0, 0);
          __builtin_amdgcn_s_wait_asynccnt(4);   // row i's 4 copies done
        } else {
          __builtin_amdgcn_s_wait_asynccnt(0);
        }
        __syncthreads();                         // publish all waves' copies
        const float a = sVal[i];
        int cidx = 0;
        for (int d = tid * 4; d < RD; d += 1024) {
          const float4 w = *(const float4*)(&wbuf[buf][d]);
          acc4[cidx].x = fmaf(a, w.x, acc4[cidx].x);
          acc4[cidx].y = fmaf(a, w.y, acc4[cidx].y);
          acc4[cidx].z = fmaf(a, w.z, acc4[cidx].z);
          acc4[cidx].w = fmaf(a, w.w, acc4[cidx].w);
          ++cidx;
        }
        __syncthreads();                         // reads done before next overwrite
      }
    }
    int cidx = 0;
    for (int d = tid * 4; d < RD; d += 1024) {
      *(float4*)(out + (size_t)b * RD + d) = acc4[cidx++];
    }
    return;
  }
#endif

  // fallback: direct-load gather
  for (int d = tid * 4; d < RD; d += 256 * 4) {
    float4 a4 = *(const float4*)(bd + d);
    for (int i = 0; i < nnz; ++i) {
      const int   h = sIdx[i];
      const float a = sVal[i];
      const float4 w = *(const float4*)(Wd + (size_t)h * RD + d);
      a4.x = fmaf(a, w.x, a4.x);
      a4.y = fmaf(a, w.y, a4.y);
      a4.z = fmaf(a, w.z, a4.z);
      a4.w = fmaf(a, w.w, a4.w);
    }
    *(float4*)(out + (size_t)b * RD + d) = a4;
  }
}

// ---------------------------------------------------------------------------
extern "C" void kernel_launch(void* const* d_in, const int* in_sizes, int n_in,
                              void* d_out, int out_size, void* d_ws, size_t ws_size,
                              hipStream_t stream) {
  const float* X  = (const float*)d_in[0];  // x_B2D      [B,2,D]
  const float* We = (const float*)d_in[1];  // W_enc_2DH  [2,D,H]
  const float* be = (const float*)d_in[2];  // b_enc_H    [H]
  const float* Wd = (const float*)d_in[3];  // W_dec_H2D  [H,2,D]
  const float* bd = (const float*)d_in[4];  // b_dec_2D   [2,D]
  const int*   kp = (const int*)d_in[5];    // k (scalar)

  const int H   = in_sizes[2];
  const int K2D = in_sizes[1] / H;          // 2*D (= 4096)
  const int B   = in_sizes[0] / K2D;        // 2048

  // workspace layout: [acts: B*H f32][hist: NBINS u32][thr: f32]
  float*    acts = (float*)d_ws;
  size_t    actsBytes = (size_t)B * H * sizeof(float);
  unsigned* hist = (unsigned*)((char*)d_ws + actsBytes);
  float*    thr  = (float*)(hist + NBINS);

  cc_zero<<<(NBINS + 255) / 256, 256, 0, stream>>>(hist, NBINS);

  dim3 egrid(H / 64, B / 64);
  cc_encode_wmma<<<egrid, 128, 0, stream>>>(X, We, be, acts, hist, B, H, K2D);

  cc_threshold<<<1, 256, 0, stream>>>(hist, kp, thr, B, H);

  cc_decode<<<B, 256, 0, stream>>>(acts, Wd, bd, thr, (float*)d_out, H, K2D);
}
